// Leroy_59485297050023
// MI455X (gfx1250) — matrix operand order repaired and checked by reading, not compiled, not run on hardware
//
#include <hip/hip_runtime.h>
#include <hip/hip_bf16.h>
#include <math.h>

// ---------------------------------------------------------------------------
// Problem constants (from reference):
//   node_groups : [4096, 1024] f32 (binary), groups_size : [1024] f32
//   out         : [4096, 4096] f32
// ---------------------------------------------------------------------------
#define N_NODES  4096
#define N_GROUPS 1024

typedef _Float16 v8h  __attribute__((ext_vector_type(8)));
typedef _Float16 v16h __attribute__((ext_vector_type(16)));
typedef float    v8f  __attribute__((ext_vector_type(8)));

// Workspace layout (bytes). Total ~112.02 MiB.
#define OFF_AS   ((size_t)0)                 // A scaled f16   : 4096*1024*2 = 8 MiB
#define OFF_BH   ((size_t)8  << 20)          // A binary f16   : 8 MiB
#define OFF_S    ((size_t)16 << 20)          // S f32          : 64 MiB
#define OFF_P    ((size_t)80 << 20)          // P f16          : 32 MiB
#define OFF_LEN  ((size_t)112 << 20)         // len f32        : 16 KiB
#define OFF_COEF (OFF_LEN + 4096 * 4)        // coef f32       : 4 KiB
#define OFF_MAX  (OFF_COEF + 1024 * 4)       // max(S) bits u32: 4 B

// ---------------------------------------------------------------------------
// Fragment loaders per CDNA5 wave32 WMMA VGPR layouts (CDNA5 ISA 7.12.2).
// A-operand 16x32 (MxK), memory row-major [rows x ldk]:
//   lane(0..15)=row m : elems 0..7 <- K k0+0..7,   elems 8..15 <- K k0+16..23
//   lane(16..31)=row m: elems 0..7 <- K k0+8..15,  elems 8..15 <- K k0+24..31
// B-operand 32x16 (KxN) taken as X^T of row-major X [cols x ldk]:
//   lane(0..15)=col n : elems 0..7 <- K k0+0..7,   elems 8..15 <- K k0+16..23
//   lane(16..31)=col n: elems 0..7 <- K k0+8..15,  elems 8..15 <- K k0+24..31
// Both paths are pure contiguous b128 loads -> no transpose loads needed.
// ---------------------------------------------------------------------------
__device__ __forceinline__ v16h load_frag_A(const _Float16* __restrict__ p, int ldk,
                                            int row0, int k0, int lm, int half) {
  const _Float16* q = p + (size_t)(row0 + lm) * ldk + k0 + half * 8;
  v8h lo = *(const v8h*)(q);
  v8h hi = *(const v8h*)(q + 16);
  v16h r;
#pragma unroll
  for (int e = 0; e < 8; ++e) { r[e] = lo[e]; r[e + 8] = hi[e]; }
  return r;
}

__device__ __forceinline__ v16h load_frag_B(const _Float16* __restrict__ p, int ldk,
                                            int col0, int k0, int lm, int half) {
  const _Float16* q = p + (size_t)(col0 + lm) * ldk + k0 + half * 8;
  v8h lo = *(const v8h*)(q);
  v8h hi = *(const v8h*)(q + 16);
  v16h r;
#pragma unroll
  for (int e = 0; e < 8; ++e) { r[e] = lo[e]; r[e + 8] = hi[e]; }
  return r;
}

__device__ __forceinline__ v8f wmma_f16(v16h a, v16h b, v8f c) {
  return __builtin_amdgcn_wmma_f32_16x16x32_f16(false, a, false, b, (short)0, c,
                                                false, false);
}

// ---------------------------------------------------------------------------
// Prep kernels
// ---------------------------------------------------------------------------
__global__ void leroy_coef_kernel(const float* __restrict__ gs, float* __restrict__ coef) {
  int g = blockIdx.x * blockDim.x + threadIdx.x;
  if (g < N_GROUPS) coef[g] = 1.0f / logf(gs[g]);
}

__global__ void leroy_pack_kernel(const float* __restrict__ ng, const float* __restrict__ coef,
                                  _Float16* __restrict__ As, _Float16* __restrict__ Bh) {
  const int total = N_NODES * N_GROUPS;
  for (int idx = blockIdx.x * blockDim.x + threadIdx.x; idx < total;
       idx += gridDim.x * blockDim.x) {
    float a = ng[idx];
    int g = idx & (N_GROUPS - 1);
    As[idx] = (_Float16)(a * coef[g]);  // coef-scaled membership (f16, <=2^-11 rel err)
    Bh[idx] = (_Float16)a;              // binary membership (exact in f16)
  }
}

__global__ __launch_bounds__(256) void leroy_rowsum_kernel(const float* __restrict__ ng,
                                                           float* __restrict__ len) {
  __shared__ float sm[256];
  int i = blockIdx.x;
  float s = 0.0f;
  for (int g = threadIdx.x; g < N_GROUPS; g += 256) s += ng[(size_t)i * N_GROUPS + g];
  sm[threadIdx.x] = s;
  __syncthreads();
  for (int w = 128; w > 0; w >>= 1) {
    if (threadIdx.x < (unsigned)w) sm[threadIdx.x] += sm[threadIdx.x + w];
    __syncthreads();
  }
  if (threadIdx.x == 0) len[i] = sm[0];
}

// ---------------------------------------------------------------------------
// GEMM1: S[i,j] = len[i]*len[j] * sum_g As[i,g]*Bh[j,g]   (K = 1024)
// Each wave computes a 64x64 tile: 4x4 WMMA accumulators, 16 v_wmma per K-step
// fed by 8 contiguous fragment loads. Fused epilogue: scale + store + max.
// ---------------------------------------------------------------------------
__global__ __launch_bounds__(256) void leroy_gemm1_kernel(
    const _Float16* __restrict__ As, const _Float16* __restrict__ Bh,
    const float* __restrict__ len, float* __restrict__ S,
    unsigned int* __restrict__ maxBits) {
  const int K = N_GROUPS;
  const int wid = (blockIdx.x * blockDim.x + threadIdx.x) >> 5;  // global wave id
  const int ty = wid >> 6;        // 64 tiles per dim (4096/64)
  const int tx = wid & 63;
  const int i0 = ty * 64, j0 = tx * 64;
  const int lane = threadIdx.x & 31;
  const int half = lane >> 4, lm = lane & 15;

  v8f acc[4][4] = {};
#pragma unroll 2
  for (int k0 = 0; k0 < K; k0 += 32) {
    v16h af[4], bf[4];
#pragma unroll
    for (int t = 0; t < 4; ++t) af[t] = load_frag_A(As, K, i0 + 16 * t, k0, lm, half);
#pragma unroll
    for (int t = 0; t < 4; ++t) bf[t] = load_frag_B(Bh, K, j0 + 16 * t, k0, lm, half);
#pragma unroll
    for (int ti = 0; ti < 4; ++ti)
#pragma unroll
      for (int tj = 0; tj < 4; ++tj) acc[ti][tj] = wmma_f16(af[ti], bf[tj], acc[ti][tj]);
  }

  // Epilogue: S = len_i * len_j * acc, track max (S >= 0).
  float lj[4];
#pragma unroll
  for (int tj = 0; tj < 4; ++tj) lj[tj] = len[j0 + 16 * tj + lm];
  float m = 0.0f;
#pragma unroll
  for (int ti = 0; ti < 4; ++ti) {
#pragma unroll
    for (int r = 0; r < 8; ++r) {
      const int row = i0 + 16 * ti + 8 * half + r;
      const float li = len[row];
#pragma unroll
      for (int tj = 0; tj < 4; ++tj) {
        float s = acc[ti][tj][r] * li * lj[tj];
        S[(size_t)row * N_NODES + j0 + 16 * tj + lm] = s;
        m = fmaxf(m, s);
      }
    }
  }
  // wave32 max-reduce, then one atomic per wave (float bits monotone for s>=0)
#pragma unroll
  for (int off = 16; off > 0; off >>= 1) m = fmaxf(m, __shfl_xor(m, off));
  if (lane == 0) atomicMax(maxBits, __float_as_uint(m));
}

// ---------------------------------------------------------------------------
// P[i,j] = log1p(S) / log1p(max(S)), stored f16 for the second WMMA GEMM.
// ---------------------------------------------------------------------------
__global__ void leroy_p_kernel(const float* __restrict__ S, _Float16* __restrict__ P,
                               const unsigned int* __restrict__ maxBits) {
  const float invml = 1.0f / log1pf(__uint_as_float(*maxBits));
  const size_t total = (size_t)N_NODES * N_NODES;
  for (size_t idx = (size_t)blockIdx.x * blockDim.x + threadIdx.x; idx < total;
       idx += (size_t)gridDim.x * blockDim.x) {
    P[idx] = (_Float16)(log1pf(S[idx]) * invml);
  }
}

// ---------------------------------------------------------------------------
// GEMM2: out = P @ P^T   (K = 4096). Same 64x64-per-wave WMMA structure.
// P (32 MiB) stays resident in the 192 MiB L2, so this is compute-bound.
// ---------------------------------------------------------------------------
__global__ __launch_bounds__(256) void leroy_gemm2_kernel(const _Float16* __restrict__ P,
                                                          float* __restrict__ out) {
  const int K = N_NODES;
  const int wid = (blockIdx.x * blockDim.x + threadIdx.x) >> 5;
  const int ty = wid >> 6;
  const int tx = wid & 63;
  const int i0 = ty * 64, j0 = tx * 64;
  const int lane = threadIdx.x & 31;
  const int half = lane >> 4, lm = lane & 15;

  v8f acc[4][4] = {};
#pragma unroll 2
  for (int k0 = 0; k0 < K; k0 += 32) {
    v16h af[4], bf[4];
#pragma unroll
    for (int t = 0; t < 4; ++t) af[t] = load_frag_A(P, K, i0 + 16 * t, k0, lm, half);
#pragma unroll
    for (int t = 0; t < 4; ++t) bf[t] = load_frag_B(P, K, j0 + 16 * t, k0, lm, half);
#pragma unroll
    for (int ti = 0; ti < 4; ++ti)
#pragma unroll
      for (int tj = 0; tj < 4; ++tj) acc[ti][tj] = wmma_f16(af[ti], bf[tj], acc[ti][tj]);
  }

#pragma unroll
  for (int ti = 0; ti < 4; ++ti) {
#pragma unroll
    for (int r = 0; r < 8; ++r) {
      const int row = i0 + 16 * ti + 8 * half + r;
#pragma unroll
      for (int tj = 0; tj < 4; ++tj) {
        float v = acc[ti][tj][r];
        v = (v == v) ? v : 0.0f;  // nan_to_num
        out[(size_t)row * N_NODES + j0 + 16 * tj + lm] = v;
      }
    }
  }
}

// ---------------------------------------------------------------------------
// Host launcher (graph-capture safe: only stream-ordered ops).
// ---------------------------------------------------------------------------
extern "C" void kernel_launch(void* const* d_in, const int* in_sizes, int n_in,
                              void* d_out, int out_size, void* d_ws, size_t ws_size,
                              hipStream_t stream) {
  const float* node_groups = (const float*)d_in[0];  // [4096,1024]
  const float* groups_size = (const float*)d_in[1];  // [1024]
  float* out = (float*)d_out;                        // [4096,4096]

  char* ws = (char*)d_ws;
  _Float16* As    = (_Float16*)(ws + OFF_AS);
  _Float16* Bh    = (_Float16*)(ws + OFF_BH);
  float*    S     = (float*)(ws + OFF_S);
  _Float16* P     = (_Float16*)(ws + OFF_P);
  float*    len   = (float*)(ws + OFF_LEN);
  float*    coef  = (float*)(ws + OFF_COEF);
  unsigned* maxBv = (unsigned*)(ws + OFF_MAX);

  // zero the atomic-max cell (ws is poisoned with 0xAA which compares huge as uint)
  hipMemsetAsync(maxBv, 0, sizeof(unsigned), stream);

  leroy_coef_kernel<<<4, 256, 0, stream>>>(groups_size, coef);
  leroy_pack_kernel<<<4096, 256, 0, stream>>>(node_groups, coef, As, Bh);
  leroy_rowsum_kernel<<<N_NODES, 256, 0, stream>>>(node_groups, len);

  // 64x64 tile per wave -> (4096/64)^2 = 4096 waves -> 512 blocks of 8 waves
  leroy_gemm1_kernel<<<512, 256, 0, stream>>>(As, Bh, len, S, maxBv);
  leroy_p_kernel<<<8192, 256, 0, stream>>>(S, P, maxBv);
  leroy_gemm2_kernel<<<512, 256, 0, stream>>>(P, out);
}
